// Mymodel_39788577030510
// MI455X (gfx1250) — matrix operand order repaired and checked by reading, not compiled, not run on hardware
//
#include <hip/hip_runtime.h>
#include <hip/hip_bf16.h>
#include <math.h>

// ---- model dims ----
#define B_   16
#define S_   512
#define H_   1024
#define M_   16      // conv channels (h layout)
#define N_   64
#define OD_  16
#define DN_  1024
#define NL_  4
#define EPS_ 1e-5f
#define ST_  8       // conv strip rows per block
#define TLS  40      // padded LDS row stride in bf16 elems (80B, 16B aligned)

typedef __attribute__((ext_vector_type(16))) __bf16 v16bf;
typedef __attribute__((ext_vector_type(8)))  float  v8f;

union FragU { uint4 u[2]; v16bf v; unsigned short s[16]; };

__device__ __forceinline__ unsigned short f2bf(float f) {
  unsigned u = __float_as_uint(f);
  u += 0x7FFFu + ((u >> 16) & 1u);          // RNE
  return (unsigned short)(u >> 16);
}
__device__ __forceinline__ float gelu_erf(float x) {
  return 0.5f * x * (1.0f + erff(x * 0.70710678118654752f));
}

// CDNA5 async global->LDS copy (ASYNCcnt-tracked), 16B per lane.
__device__ __forceinline__ void async_ld_b128(unsigned lds_byte_off, const void* gptr) {
  asm volatile("global_load_async_to_lds_b128 %0, %1, off"
               :: "v"(lds_byte_off), "v"((unsigned long long)(uintptr_t)gptr)
               : "memory");
}
#define WAIT_ASYNC_4() asm volatile("s_wait_asynccnt 0x4" ::: "memory")
#define WAIT_ASYNC_0() asm volatile("s_wait_asynccnt 0x0" ::: "memory")

// ---------------- embedding + LN0 -> h layout [B,M,S,N] ----------------
__global__ void __launch_bounds__(256) embed_ln0_kernel(
    const int* __restrict__ x, const float* __restrict__ table,
    const float* __restrict__ g, const float* __restrict__ bta,
    float* __restrict__ h_out) {
  int token = blockIdx.x;            // b*S + s
  int b = token >> 9, s = token & 511;
  int t = threadIdx.x;
  int idx = x[token];
  float v[4]; float sum = 0.f, sq = 0.f;
#pragma unroll
  for (int j = 0; j < 4; ++j) {
    int d = t * 4 + j;
    float val = (idx == 0) ? 0.f : table[(size_t)idx * H_ + d];
    v[j] = val; sum += val; sq += val * val;
  }
  __shared__ float rs[256], rq[256];
  rs[t] = sum; rq[t] = sq;
  __syncthreads();
  for (int off = 128; off > 0; off >>= 1) {
    if (t < off) { rs[t] += rs[t + off]; rq[t] += rq[t + off]; }
    __syncthreads();
  }
  float mean = rs[0] * (1.f / H_);
  float var  = rq[0] * (1.f / H_) - mean * mean;
  float inv  = rsqrtf(var + EPS_);
#pragma unroll
  for (int j = 0; j < 4; ++j) {
    int d = t * 4 + j;
    float y = (v[j] - mean) * inv * g[d] + bta[d];
    int m = d >> 6, n = d & 63;
    h_out[(((size_t)b * M_ + m) * S_ + s) * N_ + n] = y;
  }
}

// ---------------- 3x3 SAME conv, 16->16 ch, + GELU ----------------
// mode 0: out bf16 in [B,S,OD*N] GEMM layout.  mode 1: fp32 + residual, [B,M,S,N].
__global__ void __launch_bounds__(256) conv3x3_kernel(
    const float* __restrict__ in,            // [B,16,S,N]
    const float* __restrict__ w,             // [16,16,3,3]
    const float* __restrict__ bias,          // [16]
    unsigned short* __restrict__ out_qkv,    // mode 0
    const float* __restrict__ resid,         // mode 1
    float* __restrict__ out_res,             // mode 1
    int mode) {
  __shared__ float tile[M_][ST_ + 2][N_ + 2];
  __shared__ float wt[OD_][M_][3][3];
  __shared__ float bs[OD_];
  int t = threadIdx.x;
  int s0 = blockIdx.x * ST_;
  int b  = blockIdx.y;
  for (int i = t; i < OD_ * M_ * 9; i += 256) ((float*)wt)[i] = w[i];
  if (t < OD_) bs[t] = bias[t];
  const int TOT = M_ * (ST_ + 2) * (N_ + 2);      // 10560
  for (int i = t; i < TOT; i += 256) {
    int m   = i / ((ST_ + 2) * (N_ + 2));
    int rem = i % ((ST_ + 2) * (N_ + 2));
    int sl  = rem / (N_ + 2);
    int nl  = rem % (N_ + 2);
    int sg = s0 + sl - 1, ng = nl - 1;
    float val = 0.f;
    if (sg >= 0 && sg < S_ && ng >= 0 && ng < N_)
      val = in[(((size_t)b * M_ + m) * S_ + sg) * N_ + ng];
    tile[m][sl][nl] = val;
  }
  __syncthreads();
#pragma unroll 1
  for (int r = 0; r < 32; ++r) {
    int oi  = t + 256 * r;                   // OD*ST*N = 8192 outputs / block
    int od  = oi >> 9;
    int rem = oi & 511;
    int sl  = rem >> 6;
    int n   = rem & 63;
    float acc = bs[od];
#pragma unroll
    for (int m = 0; m < M_; ++m)
#pragma unroll
      for (int ds = 0; ds < 3; ++ds)
#pragma unroll
        for (int dn = 0; dn < 3; ++dn)
          acc = fmaf(tile[m][sl + ds][n + dn], wt[od][m][ds][dn], acc);
    float y = gelu_erf(acc);
    if (mode == 0) {
      out_qkv[((size_t)b * S_ + (s0 + sl)) * DN_ + od * N_ + n] = f2bf(y);
    } else {
      size_t o = (((size_t)b * M_ + od) * S_ + (s0 + sl)) * N_ + n;
      out_res[o] = y + resid[o];
    }
  }
}

// ---------------- bf16 transpose: [B][S][DN] -> [B][DN][S] ----------------
__global__ void __launch_bounds__(256) transpose_bf16_kernel(
    const unsigned short* __restrict__ src, unsigned short* __restrict__ dst) {
  __shared__ unsigned short tl[64][64 + 8];
  int b  = blockIdx.z;
  int s0 = blockIdx.y * 64, d0 = blockIdx.x * 64;
  int t = threadIdx.x;
  int r = t >> 3, c = (t & 7) * 8;
#pragma unroll
  for (int p = 0; p < 2; ++p) {
    int row = r + p * 32;   // s-dim row
    *(uint4*)&tl[row][c] = *(const uint4*)(src + ((size_t)b * S_ + s0 + row) * DN_ + d0 + c);
  }
  __syncthreads();
#pragma unroll
  for (int p = 0; p < 2; ++p) {
    int row = r + p * 32;   // d-dim row
    unsigned short tmp[8];
#pragma unroll
    for (int j = 0; j < 8; ++j) tmp[j] = tl[c + j][row];
    *(uint4*)(dst + ((size_t)b * DN_ + d0 + row) * S_ + s0 + c) = *(uint4*)tmp;
  }
}

// ---------------- WMMA fragment loaders (wave32 layouts, 16-bit) ----------------
// A 16x32: lanes 0-15 row M=lane, K chunks {0-7,16-23}; lanes 16-31 K chunks {8-15,24-31}
__device__ __forceinline__ v16bf frag_a(const unsigned short* lds, int rowBase, int lane) {
  int r  = rowBase + (lane & 15);
  int kb = (lane & 16) ? 8 : 0;
  FragU f;
  f.u[0] = *(const uint4*)(lds + r * TLS + kb);
  f.u[1] = *(const uint4*)(lds + r * TLS + kb + 16);
  return f.v;
}
// B 32x16 stored in LDS as [n][k]: lanes 0-15 K chunks {0-7,16-23}; lanes 16-31 K chunks {8-15,24-31}
__device__ __forceinline__ v16bf frag_b(const unsigned short* lds, int rowBase, int lane) {
  int r  = rowBase + (lane & 15);
  int kb = (lane & 16) ? 8 : 0;
  FragU f;
  f.u[0] = *(const uint4*)(lds + r * TLS + kb);
  f.u[1] = *(const uint4*)(lds + r * TLS + kb + 16);
  return f.v;
}

// ------ batched GEMM NT: C[M,N] = A[M,K] * B[N,K]^T (bf16 -> f32) ------
// Async double-buffered LDS staging via GLOBAL_LOAD_ASYNC_TO_LDS_B128.
// K must be a multiple of 32 and >= 64 (last K-step is peeled).
__global__ void __launch_bounds__(256) gemm_nt_bf16(
    const unsigned short* __restrict__ A, const unsigned short* __restrict__ Bm,
    float* __restrict__ C, int Mr, int Nr, int K) {
  __shared__ unsigned short la[2][128 * TLS];
  __shared__ unsigned short lb[2][128 * TLS];
  int bz = blockIdx.z;
  const unsigned short* Ab = A  + (size_t)bz * Mr * K;
  const unsigned short* Bb = Bm + (size_t)bz * Nr * K;
  float* Cb = C + (size_t)bz * Mr * Nr;
  int row0 = blockIdx.y * 128, col0 = blockIdx.x * 128;
  int t = threadIdx.x, lane = t & 31, wave = t >> 5;
  int wr = (wave >> 1) * 32;   // 4 row groups of 32
  int wc = (wave & 1) * 64;    // 2 col groups of 64
  v8f acc[2][4] = {};

  // per-thread staging slots: half a 32-elem row per thread
  int lrow = t >> 1, lhalf = (t & 1) * 16;
  unsigned laoff[2], lboff[2];
#pragma unroll
  for (int p = 0; p < 2; ++p) {
    laoff[p] = (unsigned)(uintptr_t)&la[p][lrow * TLS + lhalf];
    lboff[p] = (unsigned)(uintptr_t)&lb[p][lrow * TLS + lhalf];
  }
  const unsigned short* gA = Ab + (size_t)(row0 + lrow) * K + lhalf;
  const unsigned short* gB = Bb + (size_t)(col0 + lrow) * K + lhalf;

  auto compute_tile = [&](int bufi) {
    const unsigned short* pa = la[bufi];
    const unsigned short* pb = lb[bufi];
    v16bf fa0 = frag_a(pa, wr,      lane);
    v16bf fa1 = frag_a(pa, wr + 16, lane);
#pragma unroll
    for (int ct = 0; ct < 4; ++ct) {
      v16bf fb = frag_b(pb, wc + ct * 16, lane);
      acc[0][ct] = __builtin_amdgcn_wmma_f32_16x16x32_bf16(false, fa0, false, fb, (short)0, acc[0][ct], false, false);
      acc[1][ct] = __builtin_amdgcn_wmma_f32_16x16x32_bf16(false, fa1, false, fb, (short)0, acc[1][ct], false, false);
    }
  };

  // prologue: stage tile 0 into buffer 0 (4 async b128 per thread)
  async_ld_b128(laoff[0],      gA);
  async_ld_b128(laoff[0] + 16, gA + 8);
  async_ld_b128(lboff[0],      gB);
  async_ld_b128(lboff[0] + 16, gB + 8);

  int buf = 0;
  for (int k0 = 0; k0 < K - 32; k0 += 32) {   // steady state: branch-free issue + wait(4)
    int nb = buf ^ 1;
    const unsigned short* nA = gA + k0 + 32;
    const unsigned short* nB = gB + k0 + 32;
    async_ld_b128(laoff[nb],      nA);
    async_ld_b128(laoff[nb] + 16, nA + 8);
    async_ld_b128(lboff[nb],      nB);
    async_ld_b128(lboff[nb] + 16, nB + 8);
    WAIT_ASYNC_4();             // per-wave in-order => current tile landed
    __syncthreads();            // all waves' tiles visible
    compute_tile(buf);
    __syncthreads();            // safe to overwrite this buffer next step
    buf = nb;
  }
  // epilogue: last tile
  WAIT_ASYNC_0();
  __syncthreads();
  compute_tile(buf);

  int rOff = (lane >> 4) * 8, nOff = lane & 15;
#pragma unroll
  for (int rt = 0; rt < 2; ++rt)
#pragma unroll
    for (int ct = 0; ct < 4; ++ct)
#pragma unroll
      for (int r = 0; r < 8; ++r) {
        int row = row0 + wr + rt * 16 + rOff + r;
        int col = col0 + wc + ct * 16 + nOff;
        Cb[(size_t)row * Nr + col] = acc[rt][ct][r];
      }
}

// ---------------- softmax over the BATCH axis ----------------
__global__ void __launch_bounds__(256) softmax_batch_kernel(
    const float* __restrict__ logits, unsigned short* __restrict__ score) {
  int i = blockIdx.x * 256 + threadIdx.x;       // 0 .. S*S-1
  const int STRD = S_ * S_;
  float v[B_]; float mx = -INFINITY;
#pragma unroll
  for (int b = 0; b < B_; ++b) { v[b] = logits[(size_t)b * STRD + i]; mx = fmaxf(mx, v[b]); }
  float sum = 0.f;
#pragma unroll
  for (int b = 0; b < B_; ++b) { v[b] = __expf(v[b] - mx); sum += v[b]; }
  float inv = 1.f / sum;
#pragma unroll
  for (int b = 0; b < B_; ++b) score[(size_t)b * STRD + i] = f2bf(v[b] * inv);
}

// ---------------- LN over DN=1024 + relayout [B,S,DN] -> [B,OD,S,N] ----------------
__global__ void __launch_bounds__(256) ln_relayout_kernel(
    const float* __restrict__ o, const float* __restrict__ g,
    const float* __restrict__ bt, float* __restrict__ oc) {
  int token = blockIdx.x; int b = token >> 9, s = token & 511;
  int t = threadIdx.x;
  const float* row = o + (size_t)token * DN_;
  float v[4]; float sum = 0.f, sq = 0.f;
#pragma unroll
  for (int j = 0; j < 4; ++j) {
    int d = t * 4 + j; v[j] = row[d]; sum += v[j]; sq += v[j] * v[j];
  }
  __shared__ float rs[256], rq[256];
  rs[t] = sum; rq[t] = sq;
  __syncthreads();
  for (int off = 128; off > 0; off >>= 1) {
    if (t < off) { rs[t] += rs[t + off]; rq[t] += rq[t + off]; }
    __syncthreads();
  }
  float mean = rs[0] * (1.f / DN_);
  float var  = rq[0] * (1.f / DN_) - mean * mean;
  float inv  = rsqrtf(var + EPS_);
#pragma unroll
  for (int j = 0; j < 4; ++j) {
    int d = t * 4 + j;
    float y = (v[j] - mean) * inv * g[d] + bt[d];
    int od = d >> 6, n = d & 63;
    oc[(((size_t)b * OD_ + od) * S_ + s) * N_ + n] = y;
  }
}

extern "C" void kernel_launch(void* const* d_in, const int* in_sizes, int n_in,
                              void* d_out, int out_size, void* d_ws, size_t ws_size,
                              hipStream_t stream) {
  (void)in_sizes; (void)n_in; (void)out_size; (void)ws_size;
  const int*   x    = (const int*)d_in[0];
  const float* emb  = (const float*)d_in[1];
  const float* ln0g = (const float*)d_in[2];
  const float* ln0b = (const float*)d_in[3];
  const float* qw   = (const float*)d_in[4];
  const float* qb   = (const float*)d_in[5];
  const float* kw   = (const float*)d_in[6];
  const float* kb   = (const float*)d_in[7];
  const float* vw   = (const float*)d_in[8];
  const float* vb   = (const float*)d_in[9];
  const float* lng  = (const float*)d_in[10];
  const float* lnb  = (const float*)d_in[11];
  const float* cw   = (const float*)d_in[12];
  const float* cb   = (const float*)d_in[13];
  float* out = (float*)d_out;
  char*  ws  = (char*)d_ws;

  const size_t SL = (size_t)B_ * M_ * S_ * N_;   // 8,388,608 floats per h slice
  // workspace layout (oc aliases q2/k2, which are dead by then)
  unsigned short* q2     = (unsigned short*)(ws + 0);
  unsigned short* k2     = (unsigned short*)(ws + 16777216);
  unsigned short* v2     = (unsigned short*)(ws + 33554432);
  unsigned short* v2t    = (unsigned short*)(ws + 50331648);   // [B][DN][S]
  float*          logits = (float*)(ws + 67108864);
  unsigned short* score  = (unsigned short*)(ws + 83886080);
  float*          o      = (float*)(ws + 92274688);
  float*          oc     = (float*)(ws + 0);

  // h0 (layer-0 input) lives in d_out slice 1; layer i writes slice i+2; last layer -> slice 0
  embed_ln0_kernel<<<B_ * S_, 256, 0, stream>>>(x, emb, ln0g, ln0b, out + SL);

  for (int i = 0; i < NL_; ++i) {
    const float* hin  = out + SL * (size_t)(1 + i);
    float*       hout = (i == NL_ - 1) ? out : out + SL * (size_t)(2 + i);
    dim3 cg(S_ / ST_, B_);
    conv3x3_kernel<<<cg, 256, 0, stream>>>(hin, qw + i * 2304, qb + i * 16, q2, nullptr, nullptr, 0);
    conv3x3_kernel<<<cg, 256, 0, stream>>>(hin, kw + i * 2304, kb + i * 16, k2, nullptr, nullptr, 0);
    conv3x3_kernel<<<cg, 256, 0, stream>>>(hin, vw + i * 2304, vb + i * 16, v2, nullptr, nullptr, 0);
    transpose_bf16_kernel<<<dim3(DN_ / 64, S_ / 64, B_), 256, 0, stream>>>(v2, v2t);
    // Q·K^T  ->  logits[b][512][512]
    gemm_nt_bf16<<<dim3(S_ / 128, S_ / 128, B_), 256, 0, stream>>>(q2, k2, logits, S_, S_, DN_);
    softmax_batch_kernel<<<(S_ * S_) / 256, 256, 0, stream>>>(logits, score);
    // score·V  ->  o[b][512][1024]  (NT against transposed V)
    gemm_nt_bf16<<<dim3(DN_ / 128, S_ / 128, B_), 256, 0, stream>>>(score, v2t, o, S_, DN_, S_);
    ln_relayout_kernel<<<B_ * S_, 256, 0, stream>>>(o, lng + i * DN_, lnb + i * DN_, oc);
    conv3x3_kernel<<<cg, 256, 0, stream>>>(oc, cw + i * 2304, cb + i * 16, nullptr, hin, hout, 1);
  }
}
